// PredictingModule_7713761263922
// MI455X (gfx1250) — matrix operand order repaired and checked by reading, not compiled, not run on hardware
//
#include <hip/hip_runtime.h>
#include <math.h>

typedef _Float16 half_t;
typedef __attribute__((ext_vector_type(16))) _Float16 v16h;
typedef __attribute__((ext_vector_type(8)))  _Float16 v8h;
typedef __attribute__((ext_vector_type(4)))  _Float16 v4h;
typedef __attribute__((ext_vector_type(8)))  float    v8f;
typedef __attribute__((ext_vector_type(4)))  float    v4f;

#define WAVES_PER_BLOCK 4
#define XBUF_STRIDE 272                      // halves per x row (256 + 16 pad)
#define H1_STRIDE   136                      // halves per h1 row (128 + 8 pad)
#define XBUF_HALVES (16 * XBUF_STRIDE)       // 4352 halves = 8704 B
#define H1_HALVES   (16 * H1_STRIDE)         // 2176 halves = 4352 B
#define WAVE_HALVES (XBUF_HALVES + H1_HALVES)

// Intra-wave LDS ordering: all DS ops from this wave complete, and the
// compiler may not move LDS accesses across this point.
__device__ __forceinline__ void lds_fence() {
  asm volatile("s_wait_dscnt 0" ::: "memory");
}

// A-fragment for v_wmma_f32_16x16x32_f16 from a row-major f16 row in LDS.
// 16-bit 16x32 A layout (wave32), contiguous K order matching B:
//   lanes 0-15 : M=lane,    K = k0+{0..15}
//   lanes 16-31: M=lane-16, K = k0+{16..31}
__device__ __forceinline__ v16h load_a_frag(const half_t* row, int k0, int hi) {
  const v8h c1 = *(const v8h*)(row + k0 + hi * 16);
  const v8h c2 = *(const v8h*)(row + k0 + hi * 16 + 8);
  v16h a;
#pragma unroll
  for (int i = 0; i < 8; ++i) { a[i] = c1[i]; a[8 + i] = c2[i]; }
  return a;
}

// B-fragment: 16-bit 32x16 B layout (wave32), contiguous K order matching A:
//   lanes 0-15 : N=lane,    K = k0+0..15  (contiguous, 2 halves per VGPR)
//   lanes 16-31: N=lane-16, K = k0+16..31 (contiguous)
// With W transposed to [N, K] f16 row-major, this is one contiguous 32B load.
__device__ __forceinline__ v16h load_b_frag(const half_t* WT, int ldk,
                                            int n, int k0, int hi) {
  return *(const v16h*)(WT + (long)n * ldk + k0 + hi * 16);
}

// Convert W1 [256,128] f32 -> W1T f16 [128,256]; W2 [128,32] f32 -> W2T f16 [32,128].
__global__ void prep_weights_kernel(const float* __restrict__ W1,
                                    const float* __restrict__ W2,
                                    half_t* __restrict__ ws) {
  int tid = blockIdx.x * 256 + threadIdx.x;
  if (tid < 128 * 256) {               // W1T
    int n = tid >> 8, k = tid & 255;
    ws[n * 256 + k] = (half_t)W1[k * 128 + n];
  }
  int t2 = tid - 128 * 256;
  if (t2 >= 0 && t2 < 32 * 128) {      // W2T
    int n = t2 >> 7, k = t2 & 127;
    ws[128 * 256 + n * 128 + k] = (half_t)W2[k * 32 + n];
  }
}

__global__ __launch_bounds__(WAVES_PER_BLOCK * 32)
void mlp_predict_kernel(const float* __restrict__ h_user,
                        const float* __restrict__ h_item,
                        const int* __restrict__ src_ui, const int* __restrict__ dst_ui,
                        const int* __restrict__ src_iu, const int* __restrict__ dst_iu,
                        const float* __restrict__ b1, const float* __restrict__ b2,
                        const float* __restrict__ W3, const float* __restrict__ b3,
                        const half_t* __restrict__ W1T, const half_t* __restrict__ W2T,
                        float* __restrict__ out, int E, int nGroupsUI, int nGroups) {
  __shared__ __align__(16) half_t smem[WAVES_PER_BLOCK * WAVE_HALVES];

  const int wave = threadIdx.x >> 5;
  const int lane = threadIdx.x & 31;
  const int group = blockIdx.x * WAVES_PER_BLOCK + wave;
  if (group >= nGroups) return;

  const bool is_ui = group < nGroupsUI;
  const int  gl    = is_ui ? group : group - nGroupsUI;
  const long base  = (long)gl * 16;

  const int*   srcIdx = is_ui ? src_ui : src_iu;
  const int*   dstIdx = is_ui ? dst_ui : dst_iu;
  const float* headT  = is_ui ? h_user : h_item;   // first 128 features
  const float* tailT  = is_ui ? h_item : h_user;   // last  128 features
  float*       outp   = out + (is_ui ? 0L : (long)E) + base;

  half_t* xbuf  = smem + wave * WAVE_HALVES;       // 16 x 256 f16 (stride 272)
  half_t* h1buf = xbuf + XBUF_HALVES;              // 16 x 128 f16 (stride 136)
  float*  h2f   = (float*)xbuf;                    // reuse: 16 x 32 f32 (stride 34)

  // ---- Stage x tile = [head_row | tail_row] as f16 into LDS (coalesced B128) ----
#pragma unroll 4
  for (int i = lane; i < 16 * 64; i += 32) {
    int row = i >> 6;                 // edge within group
    int c4  = i & 63;                 // float4 column within 256-wide row
    long e  = base + row;
    int  er = (e < (long)E) ? (int)e : 0;
    int  ridx = (c4 < 32) ? srcIdx[er] : dstIdx[er];
    const float* tab = (c4 < 32) ? headT : tailT;
    v4f v = *(const v4f*)(tab + (long)ridx * 128 + (c4 & 31) * 4);
    v4h h = { (half_t)v.x, (half_t)v.y, (half_t)v.z, (half_t)v.w };
    *(v4h*)(xbuf + row * XBUF_STRIDE + c4 * 4) = h;
  }
  lds_fence();

  const int m  = lane & 15;           // A row / B column within tile
  const int hi = lane >> 4;           // half-wave selector
  const half_t* arow = xbuf + m * XBUF_STRIDE;

  // ---- Layer 1: h1 = relu(x[16,256] @ W1[256,128] + b1), f16 WMMA, f32 accum ----
#pragma unroll
  for (int nt = 0; nt < 8; ++nt) {
    const int n0 = nt * 16;
    v8f acc = {};
#pragma unroll
    for (int kt = 0; kt < 8; ++kt) {
      const int k0 = kt * 32;
      v16h a = load_a_frag(arow, k0, hi);
      v16h b = load_b_frag(W1T, 256, n0 + m, k0, hi);
      acc = __builtin_amdgcn_wmma_f32_16x16x32_f16(false, a, false, b,
                                                   (short)0, acc, false, false);
    }
    const float bv = b1[n0 + m];
#pragma unroll
    for (int r = 0; r < 8; ++r) {     // C/D: VGPR r <-> M = r + 8*hi, N = n0+m
      float v = acc[r] + bv;
      v = v > 0.0f ? v : 0.0f;
      h1buf[(r + 8 * hi) * H1_STRIDE + n0 + m] = (half_t)v;
    }
  }
  lds_fence();

  // ---- Layer 2: h2 = relu(h1[16,128] @ W2[128,32] + b2) ----
  const half_t* a2row = h1buf + m * H1_STRIDE;
#pragma unroll
  for (int nt = 0; nt < 2; ++nt) {
    const int n0 = nt * 16;
    v8f acc = {};
#pragma unroll
    for (int kt = 0; kt < 4; ++kt) {
      const int k0 = kt * 32;
      v16h a = load_a_frag(a2row, k0, hi);
      v16h b = load_b_frag(W2T, 128, n0 + m, k0, hi);
      acc = __builtin_amdgcn_wmma_f32_16x16x32_f16(false, a, false, b,
                                                   (short)0, acc, false, false);
    }
    const float bv = b2[n0 + m];
#pragma unroll
    for (int r = 0; r < 8; ++r) {
      float v = acc[r] + bv;
      v = v > 0.0f ? v : 0.0f;
      h2f[(r + 8 * hi) * 34 + n0 + m] = v;
    }
  }
  lds_fence();

  // ---- Layer 3: sigmoid(h2[16,32] @ W3[32,1] + b3), one edge per lane 0..15 ----
  if (lane < 16) {
    long e = base + lane;
    if (e < (long)E) {
      float z = b3[0];
#pragma unroll
      for (int n = 0; n < 32; ++n) z += h2f[lane * 34 + n] * W3[n];
      outp[lane] = 1.0f / (1.0f + __expf(-z));
    }
  }
}

extern "C" void kernel_launch(void* const* d_in, const int* in_sizes, int n_in,
                              void* d_out, int out_size, void* d_ws, size_t ws_size,
                              hipStream_t stream) {
  const float* h_user = (const float*)d_in[0];
  const float* h_item = (const float*)d_in[1];
  const int*   src_ui = (const int*)d_in[2];
  const int*   dst_ui = (const int*)d_in[3];
  const int*   src_iu = (const int*)d_in[4];
  const int*   dst_iu = (const int*)d_in[5];
  const float* W1     = (const float*)d_in[6];
  const float* b1     = (const float*)d_in[7];
  const float* W2     = (const float*)d_in[8];
  const float* b2     = (const float*)d_in[9];
  const float* W3     = (const float*)d_in[10];
  const float* b3     = (const float*)d_in[11];

  const int E = in_sizes[2];                 // number of edges per etype

  half_t* W1T = (half_t*)d_ws;               // [128, 256] f16 = 64 KiB
  half_t* W2T = W1T + 128 * 256;             // [ 32, 128] f16 =  8 KiB

  {
    const int total = 128 * 256 + 32 * 128;
    prep_weights_kernel<<<(total + 255) / 256, 256, 0, stream>>>(W1, W2, (half_t*)d_ws);
  }

  const int nGroupsUI = (E + 15) / 16;
  const int nGroups   = 2 * nGroupsUI;
  const int blocks    = (nGroups + WAVES_PER_BLOCK - 1) / WAVES_PER_BLOCK;

  mlp_predict_kernel<<<blocks, WAVES_PER_BLOCK * 32, 0, stream>>>(
      h_user, h_item, src_ui, dst_ui, src_iu, dst_iu,
      b1, b2, W3, b3, W1T, W2T, (float*)d_out, E, nGroupsUI, nGroups);
}
